// MyBILSTM_22119081574760
// MI455X (gfx1250) — compile-verified, matrix operand order checked by reading, not run on hardware
//
#include <hip/hip_runtime.h>
#include <hip/hip_bf16.h>

// ---------------------------------------------------------------------------
// Bidirectional LSTM for MI455X (gfx1250, wave32, WMMA, LDS-resident weights).
//   gates = [x_t ; h_{t-1}] @ W^T   as bf16 WMMA (f32 accum)
//   - 512 threads/block (16 waves = 4 waves/SIMD) for latency hiding
//   - each wave owns ONE 16x16 tile of EACH of the 4 gates for a 16-wide h
//     sub-slice -> f/i/c/o for (b,h) are acc[0..3][r] in the same lane ->
//     elementwise is pure register math (no gate buffer, no intra-step sync)
//   - explicit ping-pong double-buffering of A/B fragments
//   - per-timestep asm memory barrier prevents LICM from hoisting the
//     t-invariant weight-fragment loads (which caused scratch spilling)
//   - recurrent-weight slice (256 cols x 512 k bf16 = 256KB) staged in LDS
//     once, served by ds_load_b128 for all 256 steps
//   - ONE 8-block grid barrier per timestep (h visibility)
// ---------------------------------------------------------------------------

#define Hh   512
#define Dd   512
#define Ss   256
#define Bb   64
#define Cc   5
#define KZ   1024      // D + H
#define NG   2048      // 4 * H
#define LSTRIDE 520    // 512 + 8 pad elems: 1040B rows -> conflict-free b128

typedef unsigned short u16;
typedef __attribute__((ext_vector_type(16))) __bf16 v16bf;
typedef __attribute__((ext_vector_type(8)))  float  v8f;

__device__ __forceinline__ u16 f2bf(float f) {
    unsigned u = __builtin_bit_cast(unsigned, f);
    unsigned r = u + 0x7FFFu + ((u >> 16) & 1u);     // round-to-nearest-even
    return (u16)(r >> 16);
}
__device__ __forceinline__ float bf2f(u16 h) {
    return __builtin_bit_cast(float, ((unsigned)h) << 16);
}
__device__ __forceinline__ float sigf(float x) {
    return 1.0f / (1.0f + __expf(-x));
}

// 16-elem bf16 fragment = two 16B chunks (lanes 0-15: K{0..7,16..23},
// lanes 16-31: K{8..15,24..31}); works for global and LDS pointers.
__device__ __forceinline__ v16bf load_frag(const u16* p) {
    union { v16bf v; struct { uint4 a, b; } u; } x;
    x.u.a = *(const uint4*)p;
    x.u.b = *(const uint4*)(p + 16);
    return x.v;
}

__device__ __forceinline__ v8f wmma_bf16(v16bf a, v16bf b, v8f c) {
    return __builtin_amdgcn_wmma_f32_16x16x32_bf16(
        false, a, false, b, (short)0, c, false, false);
}

// Grid barrier (per direction, 8 blocks). Every thread fences first so each
// wave drains its own stores to device scope before signalling arrival.
__device__ __forceinline__ void grid_barrier(unsigned* cnt, unsigned* gen, unsigned n) {
    __threadfence();
    __syncthreads();
    if (threadIdx.x == 0) {
        unsigned g = __hip_atomic_load(gen, __ATOMIC_ACQUIRE, __HIP_MEMORY_SCOPE_AGENT);
        unsigned a = __hip_atomic_fetch_add(cnt, 1u, __ATOMIC_ACQ_REL, __HIP_MEMORY_SCOPE_AGENT);
        if (a + 1u == n) {
            __hip_atomic_store(cnt, 0u, __ATOMIC_RELAXED, __HIP_MEMORY_SCOPE_AGENT);
            __hip_atomic_fetch_add(gen, 1u, __ATOMIC_RELEASE, __HIP_MEMORY_SCOPE_AGENT);
        } else {
            while (__hip_atomic_load(gen, __ATOMIC_ACQUIRE, __HIP_MEMORY_SCOPE_AGENT) == g)
                __builtin_amdgcn_s_sleep(1);
        }
    }
    __syncthreads();
}

// ----------------- Kernel 1: embedding gather + fp32 -> bf16 ----------------
__global__ void embed_cast_kernel(const int* __restrict__ x,
                                  const float* __restrict__ embed,
                                  u16* __restrict__ xe) {
    int s = blockIdx.x, b = blockIdx.y;
    int idx = x[b * Ss + s];
    const float* src = embed + (size_t)idx * Dd;
    u16* dst = xe + ((size_t)s * Bb + b) * Dd;
    for (int d = threadIdx.x; d < Dd; d += blockDim.x)
        dst[d] = f2bf(src[d]);
}

// ----------------- Kernel 2: weight cast to bf16 [dir][n][k] ----------------
__global__ void wcast_kernel(const float* __restrict__ fw,
                             const float* __restrict__ bw,
                             u16* __restrict__ Wb) {
    int row = blockIdx.x;                 // 0 .. 2*NG-1
    int dir = row >> 11;
    int n   = row & (NG - 1);
    const float* src = (dir ? bw : fw) + (size_t)n * KZ;
    u16* dst = Wb + (size_t)row * KZ;
    for (int k = threadIdx.x; k < KZ; k += blockDim.x)
        dst[k] = f2bf(src[k]);
}

// ----------------- Kernel 3: persistent bidirectional recurrence ------------
// grid = (8, 2): blockIdx.y = direction, blockIdx.x = hb (64-wide h slice).
// Wave w (0..15): mtile = w&3 (batch tile), q = w>>2 (16-wide h sub-slice).
// Tile j = gate g (0..3):  n(j) = g*512 + hb*64 + q*16 + (lane&15)
__global__ __launch_bounds__(512, 1)
void lstm_recurrent_kernel(const u16* __restrict__ xe,    // [S][B][D] bf16
                           const u16* __restrict__ Wb,    // [2][NG][KZ] bf16
                           const float* __restrict__ fwd_b,
                           const float* __restrict__ bwd_b,
                           u16* __restrict__ hbuf,        // [2][2][B][H] bf16
                           u16* __restrict__ hall,        // [2][S][B][H] bf16
                           unsigned* __restrict__ bar) {  // [2][2]
    extern __shared__ u16 lds_w[];                        // [256][LSTRIDE]

    const int dir = blockIdx.y;
    const int hb  = blockIdx.x;

    const int wave  = threadIdx.x >> 5;
    const int lane  = threadIdx.x & 31;
    const int mtile = wave & 3;                 // 4 batch tiles of 16
    const int q     = wave >> 2;                // 0..3: 16-wide h sub-slice

    const int arow  = mtile * 16 + (lane & 15); // batch row for A frags
    const int koff  = (lane >> 4) * 8;          // lane-half K offset

    const u16*   Wdir = Wb + (size_t)dir * NG * KZ;
    const float* bias = dir ? bwd_b : fwd_b;
    unsigned*    bcnt = bar + dir * 2;
    unsigned*    bgen = bar + dir * 2 + 1;

    // ---- Stage recurrent-weight slice (k in [512,1024)) into LDS ----------
    // LDS row lc in [0,256): n(lc) = (lc>>6)*512 + hb*64 + (lc&63)
    for (int i = threadIdx.x; i < 256 * (Hh / 8); i += 512) {
        int lc = i >> 6;                  // 0..255  (64 chunks of 8 per row)
        int kk = (i & 63) * 8;
        int n  = (lc >> 6) * 512 + hb * 64 + (lc & 63);
        *(uint4*)&lds_w[lc * LSTRIDE + kk] =
            *(const uint4*)&Wdir[(size_t)n * KZ + Dd + kk];
    }
    __syncthreads();

    // Per-wave operand pointers + bias registers for the 4 gate tiles.
    const u16* wp[4];     // global x-part weight rows
    int        lro[4];    // LDS row offsets for h-part
    float      breg[4];   // bias[g*512 + hc]
    const int  hl = q * 16 + (lane & 15);       // h col within 64-slice
    #pragma unroll
    for (int g = 0; g < 4; ++g) {
        wp[g]   = Wdir + (size_t)(g * 512 + hb * 64 + hl) * KZ;
        lro[g]  = (g * 64 + hl) * LSTRIDE;
        breg[g] = bias[g * 512 + hb * 64 + hl];
    }

    for (int t = 0; t < Ss; ++t) {
        // Opt barrier: forbid LICM from hoisting the t-invariant weight
        // fragment loads out of the timestep loop (round-4 scratch spills).
        asm volatile("" ::: "memory");

        const int s = dir ? (Ss - 1 - t) : t;
        const u16* hprev = hbuf + ((size_t)(dir * 2 + (t & 1)) * Bb) * Hh;
        const u16* xrow  = xe + ((size_t)s * Bb + arow) * Dd;
        const u16* hrow  = hprev + (size_t)arow * Hh;

        v8f acc[4];
        #pragma unroll
        for (int g = 0; g < 4; ++g) acc[g] = v8f{};

        v16bf a0, a1, bb0[4], bb1[4];

        // ---- x part: K in [0,512), B from global (L2-resident) ------------
        // Software-pipelined: stage (k0+32) issued before consuming stage k0.
        a0 = load_frag(xrow + koff);
        #pragma unroll
        for (int g = 0; g < 4; ++g) bb0[g] = load_frag(wp[g] + koff);

        #pragma unroll
        for (int k0 = 0; k0 < Dd; k0 += 64) {
            a1 = load_frag(xrow + k0 + 32 + koff);
            #pragma unroll
            for (int g = 0; g < 4; ++g) bb1[g] = load_frag(wp[g] + k0 + 32 + koff);
            #pragma unroll
            for (int g = 0; g < 4; ++g) acc[g] = wmma_bf16(a0, bb0[g], acc[g]);

            if (k0 + 64 < Dd) {
                a0 = load_frag(xrow + k0 + 64 + koff);
                #pragma unroll
                for (int g = 0; g < 4; ++g) bb0[g] = load_frag(wp[g] + k0 + 64 + koff);
            } else {
                a0 = load_frag(hrow + koff);      // prefetch first h-part A
            }
            #pragma unroll
            for (int g = 0; g < 4; ++g) acc[g] = wmma_bf16(a1, bb1[g], acc[g]);
        }

        // ---- h part: K in [512,1024), B from LDS, A prefetched ------------
        #pragma unroll
        for (int k0 = 0; k0 < Hh; k0 += 64) {
            a1 = load_frag(hrow + k0 + 32 + koff);
            #pragma unroll
            for (int g = 0; g < 4; ++g)
                bb0[g] = load_frag(&lds_w[lro[g] + k0 + koff]);
            #pragma unroll
            for (int g = 0; g < 4; ++g) acc[g] = wmma_bf16(a0, bb0[g], acc[g]);

            if (k0 + 64 < Hh)
                a0 = load_frag(hrow + k0 + 64 + koff);
            #pragma unroll
            for (int g = 0; g < 4; ++g)
                bb1[g] = load_frag(&lds_w[lro[g] + k0 + 32 + koff]);
            #pragma unroll
            for (int g = 0; g < 4; ++g) acc[g] = wmma_bf16(a1, bb1[g], acc[g]);
        }

        // ---- in-register elementwise gate math -----------------------------
        // Lane holds, for r in {0..7}:
        //   b  = mtile*16 + 8*(lane>>4) + r,  hc = hb*64 + q*16 + (lane&15)
        //   f = acc[0][r], i = acc[1][r], c~ = acc[2][r], o = acc[3][r]
        u16* hnew   = hbuf + ((size_t)(dir * 2 + ((t + 1) & 1)) * Bb) * Hh;
        u16* hall_s = hall + (((size_t)dir * Ss + s) * Bb) * Hh;
        const int rbase = mtile * 16 + ((lane >> 4) << 3);
        const int hc    = hb * 64 + hl;
        #pragma unroll
        for (int r = 0; r < 8; ++r) {
            float f  = sigf (acc[0][r] + breg[0]);
            float ig = sigf (acc[1][r] + breg[1]);
            float ct = tanhf(acc[2][r] + breg[2]);
            float o  = sigf (acc[3][r] + breg[3]);
            float cn = (f + ig) * ct;                 // faithful to source
            float hn = o * tanhf(cn);
            u16 hb16 = f2bf(hn);
            const size_t bi = (size_t)(rbase + r) * Hh + hc;
            hnew[bi]   = hb16;
            hall_s[bi] = hb16;
        }

        // ---- single grid barrier per step (h visibility, 8 blocks) ---------
        grid_barrier(bcnt, bgen, 8);
    }
}

// ----------------- Kernel 4: pool over time + classifier --------------------
__global__ __launch_bounds__(256)
void finalize_kernel(const u16* __restrict__ hall,
                     const float* __restrict__ Wout,
                     const float* __restrict__ bout,
                     float* __restrict__ out) {
    __shared__ float pooled[Hh];
    __shared__ float red[256];
    const int b = blockIdx.x;

    for (int h = threadIdx.x; h < Hh; h += 256) {
        float m = -1e30f;
        for (int s = 0; s < Ss; ++s) {
            float hf  = bf2f(hall[(((size_t)0 * Ss + s) * Bb + b) * Hh + h]);
            float hbv = bf2f(hall[(((size_t)1 * Ss + s) * Bb + b) * Hh + h]);
            m = fmaxf(m, tanhf(hf * hbv));
        }
        pooled[h] = m;
    }
    __syncthreads();

    for (int c = 0; c < Cc; ++c) {
        float p = 0.0f;
        for (int h = threadIdx.x; h < Hh; h += 256)
            p += pooled[h] * Wout[(size_t)c * Hh + h];
        red[threadIdx.x] = p;
        __syncthreads();
        for (int st = 128; st > 0; st >>= 1) {
            if (threadIdx.x < st) red[threadIdx.x] += red[threadIdx.x + st];
            __syncthreads();
        }
        if (threadIdx.x == 0) out[(size_t)b * Cc + c] = red[0] + bout[c];
        __syncthreads();
    }
}

// ---------------------------------------------------------------------------
extern "C" void kernel_launch(void* const* d_in, const int* in_sizes, int n_in,
                              void* d_out, int out_size, void* d_ws, size_t ws_size,
                              hipStream_t stream) {
    const int*   x     = (const int*)  d_in[0];
    const float* embed = (const float*)d_in[1];
    const float* fwd_W = (const float*)d_in[2];
    const float* fwd_b = (const float*)d_in[3];
    const float* bwd_W = (const float*)d_in[4];
    const float* bwd_b = (const float*)d_in[5];
    const float* Wout  = (const float*)d_in[6];
    const float* bout  = (const float*)d_in[7];
    float* out = (float*)d_out;

    // Workspace layout
    char* ws = (char*)d_ws;
    size_t off = 0;
    unsigned* bar = (unsigned*)(ws + off);       off += 256;
    u16* xe    = (u16*)(ws + off);               off += (size_t)Ss * Bb * Dd * 2;      // 16 MB
    u16* Wb    = (u16*)(ws + off);               off += (size_t)2 * NG * KZ * 2;       // 8 MB
    u16* hbuf  = (u16*)(ws + off);               off += (size_t)2 * 2 * Bb * Hh * 2;   // 0.25 MB
    u16* hall  = (u16*)(ws + off);               off += (size_t)2 * Ss * Bb * Hh * 2;  // 32 MB
    (void)off; (void)ws_size; (void)in_sizes; (void)n_in; (void)out_size;

    // Zero barrier state and h0 buffers every launch (ws poisoned once).
    hipMemsetAsync(bar, 0, 256, stream);
    hipMemsetAsync(hbuf, 0, (size_t)2 * 2 * Bb * Hh * 2, stream);

    embed_cast_kernel<<<dim3(Ss, Bb), 256, 0, stream>>>(x, embed, xe);
    wcast_kernel<<<2 * NG, 256, 0, stream>>>(fwd_W, bwd_W, Wb);

    const size_t lds_bytes = (size_t)256 * LSTRIDE * sizeof(u16);  // 266,240 B < 320KB
    lstm_recurrent_kernel<<<dim3(8, 2), 512, lds_bytes, stream>>>(
        xe, Wb, fwd_b, bwd_b, hbuf, hall, bar);

    finalize_kernel<<<Bb, 256, 0, stream>>>(hall, Wout, bout, out);
}
